// GraphModule_2937757631006
// MI455X (gfx1250) — matrix operand order repaired and checked by reading, not compiled, not run on hardware
//
#include <hip/hip_runtime.h>

#define B_   8
#define S_   512
#define D_   768
#define H_   12
#define DK_  64
#define DFF_ 3072
#define L_   6

typedef __bf16 bf16;
typedef __attribute__((ext_vector_type(16))) __bf16 v16bf;
typedef __attribute__((ext_vector_type(8)))  __bf16 v8bf;
typedef __attribute__((ext_vector_type(8)))  float  v8f;

union V16 { v16bf v; v8bf h[2]; };
union V32B { uint4 u[4]; bf16 e[32]; };

__device__ __forceinline__ bf16 f2bf(float f) { return (bf16)f; }

// ---- CDNA5 async global->LDS copy (ASYNCcnt-tracked, bypasses VGPRs) ----
__device__ __forceinline__ void async_b128(const void* g, void* l) {
  unsigned lds = (unsigned)(unsigned long long)l;       // generic LDS ptr: low 32b = LDS offset
  unsigned long long ga = (unsigned long long)g;
  asm volatile("global_load_async_to_lds_b128 %0, %1, off"
               :: "v"(lds), "v"(ga) : "memory");
}
__device__ __forceinline__ void wait_async0() {
  asm volatile("s_wait_asynccnt 0x0" ::: "memory");
}

// ---------------- embedding gather: h[b,s,:] = embed_w[id] ----------------
__global__ void embed_kernel(const int* __restrict__ ids,
                             const float* __restrict__ ew,
                             float* __restrict__ h) {
  int row = blockIdx.x;
  int id = ids[row];
  const float* src = ew + (size_t)id * D_;
  float* dst = h + (size_t)row * D_;
  for (int j = threadIdx.x; j < D_; j += blockDim.x) dst[j] = src[j];
}

// ---------------- T5 relative position bias table [H,S,S] ----------------
__global__ void bias_kernel(const float* __restrict__ relw, float* __restrict__ bias) {
  int idx = blockIdx.x * blockDim.x + threadIdx.x;
  if (idx >= S_ * S_) return;
  int q = idx >> 9, k = idx & (S_ - 1);
  int rel = k - q;                       // memory - query
  int ret = rel > 0 ? 16 : 0;
  int n = rel < 0 ? -rel : rel;
  int val;
  if (n < 8) {
    val = n;
  } else {
    float lf = logf((float)n * 0.125f);  // log(n/8)
    int vl = 8 + (int)(lf * (8.0f / 2.772588722239781f)); // /log(16)
    val = vl < 15 ? vl : 15;
  }
  int bucket = ret + val;
  for (int hh = 0; hh < H_; hh++)
    bias[((size_t)hh * S_ + q) * S_ + k] = relw[bucket * H_ + hh];
}

// ---------------- T5 RMSNorm: out = w * x * rsqrt(mean(x^2)+eps) ----------------
__global__ void rmsnorm_kernel(const float* __restrict__ h, const float* __restrict__ w,
                               bf16* __restrict__ outb, float* __restrict__ outf,
                               int f32out) {
  __shared__ float red[256];
  int row = blockIdx.x;
  const float* x = h + (size_t)row * D_;
  float v0 = x[threadIdx.x], v1 = x[threadIdx.x + 256], v2 = x[threadIdx.x + 512];
  red[threadIdx.x] = v0 * v0 + v1 * v1 + v2 * v2;
  __syncthreads();
  for (int s = 128; s > 0; s >>= 1) {
    if (threadIdx.x < s) red[threadIdx.x] += red[threadIdx.x + s];
    __syncthreads();
  }
  float inv = rsqrtf(red[0] * (1.0f / D_) + 1e-6f);
  float vals[3] = {v0, v1, v2};
  #pragma unroll
  for (int j = 0; j < 3; j++) {
    int col = threadIdx.x + j * 256;
    float vv = w[col] * vals[j] * inv;
    if (f32out) outf[(size_t)row * D_ + col] = vv;
    else        outb[(size_t)row * D_ + col] = f2bf(vv);
  }
}

// ---------------- WMMA GEMM: C[M,N] = A[M,K](bf16) * W[N,K](f32->bf16)^T ----------------
// Double-buffered LDS; A tile staged with async global->LDS, W tile converted in-flight.
// mode 0: store bf16 scattered to [B,H,S,DK]   (q/k/v projections)
// mode 1: store bf16 relu to [M,N]             (FFN in)
// mode 2: f32 residual accumulate outf[m,n]+=v (O-proj, FFN out)
__global__ __launch_bounds__(256) void gemm_bf16(
    const bf16* __restrict__ A, const float* __restrict__ W,
    int M, int N, int K, int mode,
    bf16* __restrict__ outb, float* __restrict__ outf) {
  __shared__ alignas(16) bf16 Ash[2][128][32];
  __shared__ alignas(16) bf16 Wsh[2][128][32];
  int tid = threadIdx.x;
  int wave = tid >> 5, lane = tid & 31;
  int wm = wave >> 2, wn = wave & 3;       // 2 x 4 wave grid -> 64x32 per wave
  int lanelo = lane & 15, kh = lane >> 4;
  int m0 = blockIdx.x * 128, n0 = blockIdx.y * 128;

  v8f acc[4][2] = {};

  int row = tid >> 1, half = tid & 1;      // staging: 2 threads per tile row
  const bf16*  aptr = A + (size_t)(m0 + row) * K + half * 16;
  const float* wptr = W + (size_t)(n0 + row) * K + half * 16;

  // ---- prologue: stage tile 0 ----
  async_b128(aptr,     &Ash[0][row][half * 16]);
  async_b128(aptr + 8, &Ash[0][row][half * 16 + 8]);
  {
    const float4* wp = (const float4*)wptr;
    float4 w0 = wp[0], w1 = wp[1], w2 = wp[2], w3 = wp[3];
    v8bf p0, p1;
    p0[0]=f2bf(w0.x); p0[1]=f2bf(w0.y); p0[2]=f2bf(w0.z); p0[3]=f2bf(w0.w);
    p0[4]=f2bf(w1.x); p0[5]=f2bf(w1.y); p0[6]=f2bf(w1.z); p0[7]=f2bf(w1.w);
    p1[0]=f2bf(w2.x); p1[1]=f2bf(w2.y); p1[2]=f2bf(w2.z); p1[3]=f2bf(w2.w);
    p1[4]=f2bf(w3.x); p1[5]=f2bf(w3.y); p1[6]=f2bf(w3.z); p1[7]=f2bf(w3.w);
    *(v8bf*)&Wsh[0][row][half * 16]     = p0;
    *(v8bf*)&Wsh[0][row][half * 16 + 8] = p1;
  }

  int nk = K >> 5;
  for (int i = 0; i < nk; i++) {
    int cur = i & 1, nxt = cur ^ 1;
    wait_async0();        // our async A(i) landed in LDS
    __syncthreads();      // everyone's tile i visible; all compute(i-1) done

    bool hasnext = (i + 1) < nk;
    float4 w0, w1, w2, w3;
    if (hasnext) {
      int kb = (i + 1) << 5;
      // issue next A tile async (hidden behind compute below)
      async_b128(aptr + kb,     &Ash[nxt][row][half * 16]);
      async_b128(aptr + kb + 8, &Ash[nxt][row][half * 16 + 8]);
      // issue next W tile global loads (consumed after compute)
      const float4* wp = (const float4*)(wptr + kb);
      w0 = wp[0]; w1 = wp[1]; w2 = wp[2]; w3 = wp[3];
    }

    // ---- compute tile i ----
    V16 af[4];
    #pragma unroll
    for (int mi = 0; mi < 4; mi++) {
      int r = wm * 64 + mi * 16 + lanelo;
      af[mi].h[0] = *(const v8bf*)&Ash[cur][r][kh * 8];
      af[mi].h[1] = *(const v8bf*)&Ash[cur][r][16 + kh * 8];
    }
    #pragma unroll
    for (int nj = 0; nj < 2; nj++) {
      int c = wn * 32 + nj * 16 + lanelo;
      V16 bfv;
      bfv.h[0] = *(const v8bf*)&Wsh[cur][c][kh * 8];
      bfv.h[1] = *(const v8bf*)&Wsh[cur][c][16 + kh * 8];
      #pragma unroll
      for (int mi = 0; mi < 4; mi++)
        acc[mi][nj] = __builtin_amdgcn_wmma_f32_16x16x32_bf16(
            false, af[mi].v, false, bfv.v, (short)0, acc[mi][nj], false, false);
    }

    if (hasnext) {  // convert + stage next W tile (latency was hidden by WMMAs)
      v8bf p0, p1;
      p0[0]=f2bf(w0.x); p0[1]=f2bf(w0.y); p0[2]=f2bf(w0.z); p0[3]=f2bf(w0.w);
      p0[4]=f2bf(w1.x); p0[5]=f2bf(w1.y); p0[6]=f2bf(w1.z); p0[7]=f2bf(w1.w);
      p1[0]=f2bf(w2.x); p1[1]=f2bf(w2.y); p1[2]=f2bf(w2.z); p1[3]=f2bf(w2.w);
      p1[4]=f2bf(w3.x); p1[5]=f2bf(w3.y); p1[6]=f2bf(w3.z); p1[7]=f2bf(w3.w);
      *(v8bf*)&Wsh[nxt][row][half * 16]     = p0;
      *(v8bf*)&Wsh[nxt][row][half * 16 + 8] = p1;
    }
  }

  // epilogue (C layout: VGPR r -> row r + 8*kh, col = lanelo within 16-tile)
  #pragma unroll
  for (int mi = 0; mi < 4; mi++)
    #pragma unroll
    for (int nj = 0; nj < 2; nj++)
      #pragma unroll
      for (int r = 0; r < 8; r++) {
        int gm = m0 + wm * 64 + mi * 16 + r + 8 * kh;
        int gn = n0 + wn * 32 + nj * 16 + lanelo;
        float v = acc[mi][nj][r];
        if (mode == 0) {
          int b = gm >> 9, s = gm & 511;
          int head = gn >> 6, dk = gn & 63;
          outb[((((size_t)b * H_ + head) * S_ + s) << 6) + dk] = f2bf(v);
        } else if (mode == 1) {
          outb[(size_t)gm * N + gn] = f2bf(v > 0.f ? v : 0.f);
        } else {
          outf[(size_t)gm * N + gn] += v;
        }
      }
}

// ---------------- flash attention: per (b, h, 64-query tile) ----------------
// K chunk: async global->LDS double buffer. V chunk: global->regs prefetch,
// scattered transposed into LDS after compute (needs element transpose).
__global__ __launch_bounds__(128) void attn_kernel(
    const bf16* __restrict__ Q, const bf16* __restrict__ Kk, const bf16* __restrict__ V,
    const float* __restrict__ bias, const float* __restrict__ mask,
    bf16* __restrict__ ctx) {
  __shared__ alignas(16) bf16 Ksh[2][64][64];
  __shared__ alignas(16) bf16 VTsh[2][64][64];
  __shared__ alignas(16) bf16 Psh[4][16][64];
  int tid = threadIdx.x;
  int wave = tid >> 5, lane = tid & 31;
  int lanelo = lane & 15, kh = lane >> 4;
  int qb = blockIdx.x, hh = blockIdx.y, bb = blockIdx.z;

  const size_t headoff = ((size_t)bb * H_ + hh) * S_ * DK_;

  // Q fragments for this wave's 16 query rows (A layout, 2 frags over dk=64)
  int qrow = qb * 64 + wave * 16 + lanelo;
  const bf16* qptr = Q + headoff + (size_t)qrow * DK_;
  V16 qf[2];
  #pragma unroll
  for (int kk = 0; kk < 2; kk++) {
    qf[kk].h[0] = *(const v8bf*)(qptr + kk * 32 + kh * 8);
    qf[kk].h[1] = *(const v8bf*)(qptr + kk * 32 + 16 + kh * 8);
  }

  v8f o[4] = {};
  float mrow[8], lrow[8];
  #pragma unroll
  for (int r = 0; r < 8; r++) { mrow[r] = -1e30f; lrow[r] = 0.f; }

  int key2 = tid >> 1, halfc = tid & 1;    // staging coords: 2 threads per key row
  const bf16* kbase = Kk + headoff + (size_t)key2 * DK_ + halfc * 32;
  const bf16* vbase = V  + headoff + (size_t)key2 * DK_ + halfc * 32;

  // ---- prologue: stage chunk 0 ----
  #pragma unroll
  for (int j = 0; j < 4; j++)
    async_b128(kbase + j * 8, &Ksh[0][key2][halfc * 32 + j * 8]);
  {
    V32B vr;
    const uint4* vp = (const uint4*)vbase;
    vr.u[0] = vp[0]; vr.u[1] = vp[1]; vr.u[2] = vp[2]; vr.u[3] = vp[3];
    #pragma unroll
    for (int e = 0; e < 32; e++) VTsh[0][halfc * 32 + e][key2] = vr.e[e];
  }

  for (int kc = 0; kc < S_ / 64; kc++) {
    int cur = kc & 1, nxt = cur ^ 1;
    wait_async0();        // our async K(kc) landed
    __syncthreads();      // K/VT chunk kc visible; all compute(kc-1) done

    bool hasnext = (kc + 1) < (S_ / 64);
    V32B vr;
    if (hasnext) {
      size_t coff = (size_t)(kc + 1) * 64 * DK_;
      #pragma unroll
      for (int j = 0; j < 4; j++)
        async_b128(kbase + coff + j * 8, &Ksh[nxt][key2][halfc * 32 + j * 8]);
      const uint4* vp = (const uint4*)(vbase + coff);
      vr.u[0] = vp[0]; vr.u[1] = vp[1]; vr.u[2] = vp[2]; vr.u[3] = vp[3];
    }

    // scores S = Q * K^T  (16 x 64 per wave)
    v8f sf[4] = {};
    #pragma unroll
    for (int kk = 0; kk < 2; kk++)
      #pragma unroll
      for (int nj = 0; nj < 4; nj++) {
        V16 bfv;
        bfv.h[0] = *(const v8bf*)&Ksh[cur][nj * 16 + lanelo][kk * 32 + kh * 8];
        bfv.h[1] = *(const v8bf*)&Ksh[cur][nj * 16 + lanelo][kk * 32 + 16 + kh * 8];
        sf[nj] = __builtin_amdgcn_wmma_f32_16x16x32_bf16(
            false, qf[kk].v, false, bfv.v, (short)0, sf[nj], false, false);
      }

    // + relative position bias + key mask
    int qrow0 = qb * 64 + wave * 16 + 8 * kh;  // + r
    #pragma unroll
    for (int nj = 0; nj < 4; nj++) {
      int key = kc * 64 + nj * 16 + lanelo;
      float mterm = (1.0f - mask[bb * S_ + key]) * -1e9f;
      const float* bptr = bias + ((size_t)hh * S_ + qrow0) * S_ + key;
      #pragma unroll
      for (int r = 0; r < 8; r++) sf[nj][r] += bptr[(size_t)r * S_] + mterm;
    }

    // chunk row-max (reduce over 64 keys: 4 frags + xor-shuffle within half-wave)
    float cm[8];
    #pragma unroll
    for (int r = 0; r < 8; r++)
      cm[r] = fmaxf(fmaxf(sf[0][r], sf[1][r]), fmaxf(sf[2][r], sf[3][r]));
    #pragma unroll
    for (int off = 1; off < 16; off <<= 1)
      #pragma unroll
      for (int r = 0; r < 8; r++) cm[r] = fmaxf(cm[r], __shfl_xor(cm[r], off, 16));

    // online softmax update
    float rs[8];
    #pragma unroll
    for (int r = 0; r < 8; r++) {
      float nm = fmaxf(mrow[r], cm[r]);
      float sc = __expf(mrow[r] - nm);
      mrow[r] = nm;
      lrow[r] *= sc;
      #pragma unroll
      for (int nj = 0; nj < 4; nj++) o[nj][r] *= sc;
      rs[r] = 0.f;
    }
    #pragma unroll
    for (int nj = 0; nj < 4; nj++)
      #pragma unroll
      for (int r = 0; r < 8; r++) {
        float pv = __expf(sf[nj][r] - mrow[r]);
        sf[nj][r] = pv;
        rs[r] += pv;
      }
    #pragma unroll
    for (int off = 1; off < 16; off <<= 1)
      #pragma unroll
      for (int r = 0; r < 8; r++) rs[r] += __shfl_xor(rs[r], off, 16);
    #pragma unroll
    for (int r = 0; r < 8; r++) lrow[r] += rs[r];

    // P: C-layout -> A-layout via per-wave LDS bounce (same-wave DS ops are in-order)
    #pragma unroll
    for (int nj = 0; nj < 4; nj++)
      #pragma unroll
      for (int r = 0; r < 8; r++)
        Psh[wave][r + 8 * kh][nj * 16 + lanelo] = f2bf(sf[nj][r]);
    asm volatile("s_wait_dscnt 0x0" ::: "memory");

    // O += P * V
    #pragma unroll
    for (int kk = 0; kk < 2; kk++) {
      V16 pa;
      pa.h[0] = *(const v8bf*)&Psh[wave][lanelo][kk * 32 + kh * 8];
      pa.h[1] = *(const v8bf*)&Psh[wave][lanelo][kk * 32 + 16 + kh * 8];
      #pragma unroll
      for (int nj = 0; nj < 4; nj++) {
        V16 vb;
        vb.h[0] = *(const v8bf*)&VTsh[cur][nj * 16 + lanelo][kk * 32 + kh * 8];
        vb.h[1] = *(const v8bf*)&VTsh[cur][nj * 16 + lanelo][kk * 32 + 16 + kh * 8];
        o[nj] = __builtin_amdgcn_wmma_f32_16x16x32_bf16(
            false, pa.v, false, vb.v, (short)0, o[nj], false, false);
      }
    }

    if (hasnext) {  // scatter next V chunk transposed (latency hidden by compute)
      #pragma unroll
      for (int e = 0; e < 32; e++) VTsh[nxt][halfc * 32 + e][key2] = vr.e[e];
    }
  }

  // normalize + store ctx[b, s, h*64+dk] (bf16, GEMM-ready [M,K] layout)
  #pragma unroll
  for (int r = 0; r < 8; r++) {
    float inv = 1.0f / lrow[r];
    int srow = qb * 64 + wave * 16 + r + 8 * kh;
    bf16* cp = ctx + ((size_t)bb * S_ + srow) * (H_ * DK_) + hh * DK_;
    #pragma unroll
    for (int nj = 0; nj < 4; nj++)
      cp[nj * 16 + lanelo] = f2bf(o[nj][r] * inv);
  }
}

// ---------------- host: orchestrate the full 6-layer forward ----------------
extern "C" void kernel_launch(void* const* d_in, const int* in_sizes, int n_in,
                              void* d_out, int out_size, void* d_ws, size_t ws_size,
                              hipStream_t stream) {
  const int*   ids   = (const int*)  d_in[0];
  const float* amask = (const float*)d_in[1];
  const float* ew    = (const float*)d_in[2];
  const float* ln1   = (const float*)d_in[3];
  const float* qw    = (const float*)d_in[4];
  const float* kw    = (const float*)d_in[5];
  const float* vw    = (const float*)d_in[6];
  const float* relw  = (const float*)d_in[7];
  const float* ow    = (const float*)d_in[8];
  const float* ln2   = (const float*)d_in[9];
  const float* wiw   = (const float*)d_in[10];
  const float* wow   = (const float*)d_in[11];
  const float* flnw  = (const float*)d_in[12];
  (void)in_sizes; (void)n_in; (void)out_size; (void)ws_size;

  char* base = (char*)d_ws;
  size_t off = 0;
  auto carve = [&](size_t bytes) -> void* {
    void* r = base + off;
    off += (bytes + 255) & ~(size_t)255;
    return r;
  };
  float* hbuf  = (float*)carve((size_t)B_ * S_ * D_ * 4);      // residual stream f32
  bf16*  xbf   = (bf16*) carve((size_t)B_ * S_ * D_ * 2);      // normed activations
  bf16*  qbf   = (bf16*) carve((size_t)B_ * S_ * D_ * 2);      // [B,H,S,DK]
  bf16*  kbf   = (bf16*) carve((size_t)B_ * S_ * D_ * 2);
  bf16*  vbf   = (bf16*) carve((size_t)B_ * S_ * D_ * 2);
  bf16*  ctxbf = (bf16*) carve((size_t)B_ * S_ * D_ * 2);      // [B,S,H*DK]
  bf16*  ffbf  = (bf16*) carve((size_t)B_ * S_ * DFF_ * 2);    // FFN hidden
  float* biasb = (float*)carve((size_t)H_ * S_ * S_ * 4);      // [H,S,S]

  embed_kernel<<<B_ * S_, 256, 0, stream>>>(ids, ew, hbuf);
  bias_kernel<<<(S_ * S_ + 255) / 256, 256, 0, stream>>>(relw, biasb);

  for (int l = 0; l < L_; l++) {
    rmsnorm_kernel<<<B_ * S_, 256, 0, stream>>>(hbuf, ln1 + (size_t)l * D_, xbf, nullptr, 0);
    gemm_bf16<<<dim3(32, 6), 256, 0, stream>>>(xbf, qw + (size_t)l * D_ * D_, 4096, 768, 768, 0, qbf, nullptr);
    gemm_bf16<<<dim3(32, 6), 256, 0, stream>>>(xbf, kw + (size_t)l * D_ * D_, 4096, 768, 768, 0, kbf, nullptr);
    gemm_bf16<<<dim3(32, 6), 256, 0, stream>>>(xbf, vw + (size_t)l * D_ * D_, 4096, 768, 768, 0, vbf, nullptr);
    attn_kernel<<<dim3(S_ / 64, H_, B_), 128, 0, stream>>>(qbf, kbf, vbf, biasb, amask, ctxbf);
    gemm_bf16<<<dim3(32, 6), 256, 0, stream>>>(ctxbf, ow + (size_t)l * D_ * D_, 4096, 768, 768, 2, nullptr, hbuf);
    rmsnorm_kernel<<<B_ * S_, 256, 0, stream>>>(hbuf, ln2 + (size_t)l * D_, xbf, nullptr, 0);
    gemm_bf16<<<dim3(32, 24), 256, 0, stream>>>(xbf, wiw + (size_t)l * DFF_ * D_, 4096, 3072, 768, 1, ffbf, nullptr);
    gemm_bf16<<<dim3(32, 6), 256, 0, stream>>>(ffbf, wow + (size_t)l * D_ * DFF_, 4096, 768, 3072, 2, nullptr, hbuf);
  }

  rmsnorm_kernel<<<B_ * S_, 256, 0, stream>>>(hbuf, flnw, nullptr, (float*)d_out, 1);
}
